// MPNNTransform_3839700762650
// MI455X (gfx1250) — matrix operand order repaired and checked by reading, not compiled, not run on hardware
//
#include <hip/hip_runtime.h>
#include <hip/hip_bf16.h>
#include <math.h>

typedef _Float16 half_t;
typedef __attribute__((ext_vector_type(16))) _Float16 v16h;
typedef __attribute__((ext_vector_type(8)))  _Float16 v8h;
typedef __attribute__((ext_vector_type(8)))  float    v8f;

#define B_  32
#define N_  256
#define F_  8
#define H_  256
#define M_  (B_ * N_)     // 8192 rows
#define O3_ (3 * H_)      // 768
#define HF_ (H_ + F_)     // 264

// ---------------------------------------------------------------------------
// small utility kernels
// ---------------------------------------------------------------------------
__global__ void zero_f32_kernel(float* __restrict__ p, int n) {
    int i = blockIdx.x * blockDim.x + threadIdx.x;
    if (i < n) p[i] = 0.0f;
}

__global__ void convert_f16_kernel(const float* __restrict__ src,
                                   half_t* __restrict__ dst, int n) {
    int i = blockIdx.x * blockDim.x + threadIdx.x;
    if (i < n) dst[i] = (half_t)src[i];
}

// src [rows, cols] -> dst [cols, rows]  (f32 -> f16 transpose)
__global__ void convert_transpose_f16_kernel(const float* __restrict__ src,
                                             half_t* __restrict__ dst,
                                             int rows, int cols) {
    int i = blockIdx.x * blockDim.x + threadIdx.x;
    if (i < rows * cols) {
        int r = i / cols, c = i % cols;
        dst[c * rows + r] = (half_t)src[i];
    }
}

// ---------------------------------------------------------------------------
// h = tanh(jets @ W_emb + b_emb)    grid: M_ blocks, 256 threads
// ---------------------------------------------------------------------------
__global__ void embed_kernel(const float* __restrict__ jets,
                             const float* __restrict__ W_emb,
                             const float* __restrict__ b_emb,
                             float* __restrict__ h32,
                             half_t* __restrict__ h16) {
    int m = blockIdx.x;
    int hc = threadIdx.x;
    const float* j = jets + m * F_;
    float acc = b_emb[hc];
#pragma unroll
    for (int f = 0; f < F_; ++f) acc += j[f] * W_emb[f * H_ + hc];
    float v = tanhf(acc);
    h32[m * H_ + hc] = v;
    h16[m * H_ + hc] = (half_t)v;
}

// ---------------------------------------------------------------------------
// Per-batch attention collapse + msg + gmsg = msg @ Wi[:, :H].T + bi
// softmax over senders j is independent of receiver i (additive-constant
// invariance), so msg is a single H-vector per batch.
// grid: B_ blocks, 256 threads
// ---------------------------------------------------------------------------
__global__ void attn_msg_kernel(const float* __restrict__ h32,
                                const float* __restrict__ w_edge,
                                const float* __restrict__ W_msg,
                                const float* __restrict__ b_msg,
                                const float* __restrict__ Wi,
                                const float* __restrict__ bi,
                                float* __restrict__ gmsg) {
    __shared__ float red[N_];
    __shared__ float p[N_];
    __shared__ float hbar[H_];
    __shared__ float msg[H_];

    int b = blockIdx.x;
    int t = threadIdx.x;
    const float* hb = h32 + (size_t)b * N_ * H_;

    // s[t] = h[b,t,:] . w_edge
    float s = 0.0f;
    const float* hr = hb + t * H_;
    for (int k = 0; k < H_; ++k) s += hr[k] * w_edge[k];

    // softmax over nodes (block-wide)
    red[t] = s; __syncthreads();
    for (int off = 128; off > 0; off >>= 1) {
        if (t < off) red[t] = fmaxf(red[t], red[t + off]);
        __syncthreads();
    }
    float mx = red[0]; __syncthreads();
    float e = expf(s - mx);
    red[t] = e; __syncthreads();
    for (int off = 128; off > 0; off >>= 1) {
        if (t < off) red[t] += red[t + off];
        __syncthreads();
    }
    float denom = red[0]; __syncthreads();
    p[t] = e / denom; __syncthreads();

    // hbar[t] = sum_n p[n] * h[b,n,t]   (coalesced over t)
    float acc = 0.0f;
    for (int n = 0; n < N_; ++n) acc += p[n] * hb[n * H_ + t];
    hbar[t] = acc; __syncthreads();

    // msg[t] = tanh(sum_k hbar[k] * W_msg[k,t] + b_msg[t])
    float acc2 = b_msg[t];
    for (int k = 0; k < H_; ++k) acc2 += hbar[k] * W_msg[k * H_ + t];
    msg[t] = tanhf(acc2); __syncthreads();

    // gmsg[b,o] = bi[o] + sum_k msg[k] * Wi[o, k]   (o = g*256 + t)
    for (int g = 0; g < 3; ++g) {
        int o = g * H_ + t;
        const float* wrow = Wi + (size_t)o * HF_;
        float a = bi[o];
        for (int k = 0; k < H_; ++k) a += msg[k] * wrow[k];
        gmsg[b * O3_ + o] = a;
    }
}

// ---------------------------------------------------------------------------
// WMMA GEMM:  C[m,o] = sum_k A[m,k] * W[o,k]
//   A : f16 row-major [M,K]          (K multiple of 32)
//   W : f16 [O,K] ("B" column-major: each output channel contiguous in K)
// Register-blocked: each wave computes a 16x64 C tile = 4 accumulators that
// share ONE A fragment per K-step (4x A-traffic reduction, 4 independent
// v_wmma per load group to cover WMMA->WMMA hazard latency).
// MODE 0: store f32 raw (ld = O)
// MODE 1: bias + relu -> f16 store (ld = O)
// MODE 2: bias + atomicAdd into out32[b*O + col]  (b = row/256)
// O must be a multiple of 64.
// ---------------------------------------------------------------------------
template <int MODE>
__global__ void gemm_wmma_kernel(const half_t* __restrict__ A,
                                 const half_t* __restrict__ W,
                                 const float* __restrict__ bias,
                                 void* __restrict__ outp,
                                 int M, int K, int O) {
    int lane = threadIdx.x & 31;
    int wave = threadIdx.x >> 5;
    int gw   = blockIdx.x * 8 + wave;
    int tilesO = O >> 6;                    // 64-wide column groups
    int mt  = gw / tilesO;
    int otg = gw - mt * tilesO;
    if (mt * 16 >= M) return;               // wave-uniform guard (EXEC stays full)

    int hi  = lane >> 4;                    // 0: lanes 0-15, 1: lanes 16-31
    int l15 = lane & 15;
    int row = mt * 16 + l15;

    const half_t* arow  = A + (size_t)row * K + (hi << 4);            // K off 0/16
    const half_t* wbase = W + (size_t)(otg * 64 + l15) * K + (hi << 4); // K off 0/16
    const size_t  wstep = (size_t)16 * K;   // next 16-column tile

    v8f c0 = {}, c1 = {}, c2 = {}, c3 = {};
    for (int k0 = 0; k0 < K; k0 += 32) {
        // A fragment (shared by all 4 column tiles):
        //   lane<16 holds K = k0..k0+15 of its row,
        //   lane>=16 holds K = k0+16..k0+31   (contiguous, matches B)
        v8h alo = *(const v8h*)(arow + k0);
        v8h ahi = *(const v8h*)(arow + k0 + 8);
        v16h a = __builtin_shufflevector(alo, ahi,
                    0, 1, 2, 3, 4, 5, 6, 7, 8, 9, 10, 11, 12, 13, 14, 15);
        // B fragments: lanes 0-15 hold K = k0..k0+15 of their column,
        //              lanes 16-31 hold K = k0+16..k0+31 (contiguous in W[O,K])
        v16h b0 = *(const v16h*)(wbase + 0 * wstep + k0);
        v16h b1 = *(const v16h*)(wbase + 1 * wstep + k0);
        v16h b2 = *(const v16h*)(wbase + 2 * wstep + k0);
        v16h b3 = *(const v16h*)(wbase + 3 * wstep + k0);
        c0 = __builtin_amdgcn_wmma_f32_16x16x32_f16(false, a, false, b0, (short)0, c0, false, false);
        c1 = __builtin_amdgcn_wmma_f32_16x16x32_f16(false, a, false, b1, (short)0, c1, false, false);
        c2 = __builtin_amdgcn_wmma_f32_16x16x32_f16(false, a, false, b2, (short)0, c2, false, false);
        c3 = __builtin_amdgcn_wmma_f32_16x16x32_f16(false, a, false, b3, (short)0, c3, false, false);
    }

    // C layout: value i of lane -> M = mt*16 + i + 8*hi, Ncol = tile*16 + l15
    int rbase = mt * 16 + (hi << 3);
    v8f acc[4] = {c0, c1, c2, c3};
#pragma unroll
    for (int j = 0; j < 4; ++j) {
        int col = otg * 64 + j * 16 + l15;
        if (MODE == 0) {
            float* out = (float*)outp;
#pragma unroll
            for (int i = 0; i < 8; ++i)
                out[(size_t)(rbase + i) * O + col] = acc[j][i];
        } else if (MODE == 1) {
            half_t* out = (half_t*)outp;
            float bv = bias[col];
#pragma unroll
            for (int i = 0; i < 8; ++i) {
                float v = acc[j][i] + bv;
                v = v > 0.0f ? v : 0.0f;
                out[(size_t)(rbase + i) * O + col] = (half_t)v;
            }
        } else {
            float* out = (float*)outp;          // [B, O]
            float bv = bias[col];
            int bb = (mt * 16) >> 8;            // 256 rows/batch, tiles never straddle
#pragma unroll
            for (int i = 0; i < 8; ++i)
                atomicAdd(&out[bb * O + col], acc[j][i] + bv);
        }
    }
}

// ---------------------------------------------------------------------------
// GRU gate combine.  grid: M_ blocks, 256 threads
// gi[o] = gmsg[b,o] + jets[m,:] . Wi[o, H:H+F]     (msg part precomputed)
// gh[o] = gh_ws[m,o] + bh[o]
// ---------------------------------------------------------------------------
__global__ void gru_kernel(const float* __restrict__ jets,
                           const float* __restrict__ Wi,
                           const float* __restrict__ gmsg,
                           const float* __restrict__ gh_ws,
                           const float* __restrict__ bh,
                           float* __restrict__ h32,
                           half_t* __restrict__ h16) {
    int m = blockIdx.x;
    int t = threadIdx.x;
    int b = m >> 8;

    float jf[F_];
    const float* j = jets + m * F_;
#pragma unroll
    for (int f = 0; f < F_; ++f) jf[f] = j[f];

    float gi[3], gh[3];
#pragma unroll
    for (int g = 0; g < 3; ++g) {
        int o = g * H_ + t;
        float a = gmsg[b * O3_ + o];
        const float* wrow = Wi + (size_t)o * HF_ + H_;
#pragma unroll
        for (int f = 0; f < F_; ++f) a += jf[f] * wrow[f];
        gi[g] = a;
        gh[g] = gh_ws[(size_t)m * O3_ + o] + bh[o];
    }
    float r  = 1.0f / (1.0f + __expf(-(gi[0] + gh[0])));
    float z  = 1.0f / (1.0f + __expf(-(gi[1] + gh[1])));
    float nn = tanhf(gi[2] + r * gh[2]);
    float ho = h32[m * H_ + t];
    float hn = (1.0f - z) * nn + z * ho;
    h32[m * H_ + t] = hn;
    h16[m * H_ + t] = (half_t)hn;
}

// ---------------------------------------------------------------------------
extern "C" void kernel_launch(void* const* d_in, const int* in_sizes, int n_in,
                              void* d_out, int out_size, void* d_ws, size_t ws_size,
                              hipStream_t stream) {
    const float* jets   = (const float*)d_in[0];
    const float* W_emb  = (const float*)d_in[1];
    const float* b_emb  = (const float*)d_in[2];
    const float* w_edge = (const float*)d_in[3];
    /* d_in[4] = b_edge: cancels inside softmax, unused */
    const float* W_msg  = (const float*)d_in[5];
    const float* b_msg  = (const float*)d_in[6];
    const float* Wi     = (const float*)d_in[7];
    const float* bi     = (const float*)d_in[8];
    const float* Wh     = (const float*)d_in[9];
    const float* bh     = (const float*)d_in[10];
    const float* Wr1    = (const float*)d_in[11];
    const float* br1    = (const float*)d_in[12];
    const float* Wr2    = (const float*)d_in[13];
    const float* br2    = (const float*)d_in[14];
    float* out = (float*)d_out;                     // [B, H]

    // workspace carve-up (256B aligned)
    char* ws = (char*)d_ws;
    size_t off = 0;
    auto take = [&](size_t bytes) {
        char* p = ws + off;
        off += (bytes + 255) & ~(size_t)255;
        return p;
    };
    float*  h32     = (float*) take((size_t)M_ * H_ * 4);    // 8 MB
    half_t* h16     = (half_t*)take((size_t)M_ * H_ * 2);    // 4 MB
    half_t* relu16  = (half_t*)take((size_t)M_ * H_ * 2);    // 4 MB
    float*  gh_ws   = (float*) take((size_t)M_ * O3_ * 4);   // 24 MB
    float*  gmsg    = (float*) take((size_t)B_ * O3_ * 4);
    half_t* Wh16    = (half_t*)take((size_t)O3_ * H_ * 2);
    half_t* Wr1T16  = (half_t*)take((size_t)H_ * H_ * 2);
    half_t* Wr2T16  = (half_t*)take((size_t)H_ * H_ * 2);
    (void)ws_size; (void)in_sizes; (void)n_in; (void)out_size;

    // 0) zero the (poisoned) output accumulator
    zero_f32_kernel<<<(B_ * H_ + 255) / 256, 256, 0, stream>>>(out, B_ * H_);

    // 1) weight conversion: Wh is already [O,K]; Wr1/Wr2 need transpose so
    //    each output channel's K-vector is contiguous for WMMA B fragments.
    convert_f16_kernel<<<(O3_ * H_ + 255) / 256, 256, 0, stream>>>(Wh, Wh16, O3_ * H_);
    convert_transpose_f16_kernel<<<(H_ * H_ + 255) / 256, 256, 0, stream>>>(Wr1, Wr1T16, H_, H_);
    convert_transpose_f16_kernel<<<(H_ * H_ + 255) / 256, 256, 0, stream>>>(Wr2, Wr2T16, H_, H_);

    // 2) embedding
    embed_kernel<<<M_, H_, 0, stream>>>(jets, W_emb, b_emb, h32, h16);

    // 3) message-passing iterations
    const int ghBlocks = (M_ / 16) * (O3_ / 64) / 8;   // 512*12/8 = 768 blocks
    for (int it = 0; it < 3; ++it) {
        attn_msg_kernel<<<B_, N_, 0, stream>>>(h32, w_edge, W_msg, b_msg, Wi, bi, gmsg);
        gemm_wmma_kernel<0><<<ghBlocks, 256, 0, stream>>>(h16, Wh16, nullptr, gh_ws,
                                                          M_, H_, O3_);
        gru_kernel<<<M_, H_, 0, stream>>>(jets, Wi, gmsg, gh_ws, bh, h32, h16);
    }

    // 4) readout: relu(h@Wr1+br1) @ Wr2 + br2, summed over nodes into out
    const int roBlocks = (M_ / 16) * (H_ / 64) / 8;    // 512*4/8 = 256 blocks
    gemm_wmma_kernel<1><<<roBlocks, 256, 0, stream>>>(h16, Wr1T16, br1, relu16,
                                                      M_, H_, H_);
    gemm_wmma_kernel<2><<<roBlocks, 256, 0, stream>>>(relu16, Wr2T16, br2, out,
                                                      M_, H_, H_);
}
